// LinearAttention_12592844112127
// MI455X (gfx1250) — matrix operand order repaired and checked by reading, not compiled, not run on hardware
//
#include <hip/hip_runtime.h>
#include <hip/hip_bf16.h>
#include <stdint.h>

#define NUM_HEADS 12
#define HEAD_DIM  64
#define C_DIM     768
#define B_SZ      4
#define N_SEQ     8192
#define M_TOT     (B_SZ * N_SEQ)     /* 32768 */
#define C3        (3 * C_DIM)        /* 2304  */
#define BH        (B_SZ * NUM_HEADS) /* 48    */

// Block-tile geometry for the two big GEMMs (K = 768 for both)
#define BM 128
#define BN 64
#define BK 64
#define LDS_LD 72   /* BK + 8 pad; 72*2B = 144B row stride, 16B aligned, */
                    /* and (m*36)%64 visits 16 distinct banks for m=0..15 */

typedef __attribute__((ext_vector_type(16))) __bf16 v16bf;
typedef __attribute__((ext_vector_type(8)))  float  v8f;
typedef int i32x4 __attribute__((ext_vector_type(4)));

#define AS1 __attribute__((address_space(1)))
#define AS3 __attribute__((address_space(3)))

union FragU { v16bf v; uint4 q[2]; };
union Pack8 { uint4 q; __bf16 h[8]; };

// ---------------------------------------------------------------------------
// gfx1250 async global->LDS path (ASYNCcnt-tracked). Signature per hipcc
// diagnostic: (i32x4 AS1* gsrc, i32x4 AS3* ldst, imm offset, imm cpol).
// Guarded so the file still compiles on toolchains without the builtins; the
// fallback is a plain b128 load + ds_store_b128, synchronized by the barrier.
// ---------------------------------------------------------------------------
#if defined(__has_builtin)
# if __has_builtin(__builtin_amdgcn_global_load_async_to_lds_b128) && \
     __has_builtin(__builtin_amdgcn_s_wait_asynccnt)
#  define USE_ASYNC_LDS 1
# endif
#endif
#ifndef USE_ASYNC_LDS
# define USE_ASYNC_LDS 0
#endif

__device__ inline void cp16(__bf16* ldst, const __bf16* gsrc) {
#if USE_ASYNC_LDS
  __builtin_amdgcn_global_load_async_to_lds_b128(
      (AS1 i32x4*)gsrc, (AS3 i32x4*)ldst, 0, 0);
#else
  *(uint4*)ldst = *(const uint4*)gsrc;
#endif
}

__device__ inline void async_wait() {
#if USE_ASYNC_LDS
  __builtin_amdgcn_s_wait_asynccnt(0);
#endif
}

// ---------------------------------------------------------------------------
// WMMA fragment loaders (contiguous k order k = 16*(lane/16) + s, same for A and B)
// A (16x32): lane holds row m=lane&15; elements 0..15 =
//   K klo..klo+15, klo = (lane>=16)?16:0.
// B (32x16) from Bt[j][k]: lane holds col j=lane&15, elements 0..15 =
//   K kb..kb+15, kb = (lane>=16)?16:0.
// ---------------------------------------------------------------------------
__device__ inline v16bf frag_a(const __bf16* p, int ld, int lane) {
  const int m   = lane & 15;
  const int klo = (lane >> 4) << 4;
  FragU u;
  u.q[0] = *(const uint4*)(p + m * ld + klo);
  u.q[1] = *(const uint4*)(p + m * ld + klo + 8);
  return u.v;
}
__device__ inline v16bf frag_b(const __bf16* p, int ld, int lane) {
  const int j  = lane & 15;
  const int kb = (lane >> 4) << 4;
  const uint4* q = (const uint4*)(p + j * ld + kb);
  FragU u;
  u.q[0] = q[0];
  u.q[1] = q[1];
  return u.v;
}

#define WMMA_BF16(acc, a, b)                                                   \
  acc = __builtin_amdgcn_wmma_f32_16x16x32_bf16(false, (a), false, (b),        \
                                                (short)0, (acc), false, false)

__device__ inline float elu1(float x) { return x > 0.f ? x + 1.f : __expf(x); }

// ---------------------------------------------------------------------------
// LDS staging: coalesced copies (8 consecutive threads move one 128B row
// segment) of A panel (BM x BK, src ld = 768) and B panel (BN x BK).
// ---------------------------------------------------------------------------
__device__ inline void stageA(__bf16* dst, const __bf16* src, int tid) {
  #pragma unroll
  for (int c = 0; c < (BM * BK / 8) / 256; ++c) {   // 4
    const int i = c * 256 + tid;
    const int row = i >> 3, kq = i & 7;
    cp16(dst + row * LDS_LD + kq * 8, src + row * C_DIM + kq * 8);
  }
}
__device__ inline void stageB(__bf16* dst, const __bf16* src, int tid) {
  #pragma unroll
  for (int c = 0; c < (BN * BK / 8) / 256; ++c) {   // 2
    const int i = c * 256 + tid;
    const int row = i >> 3, kq = i & 7;
    cp16(dst + row * LDS_LD + kq * 8, src + row * C_DIM + kq * 8);
  }
}

// ---------------------------------------------------------------------------
// Elementwise helpers
// ---------------------------------------------------------------------------
__global__ void __launch_bounds__(256)
cvt_f32_bf16(const float* __restrict__ src, __bf16* __restrict__ dst, int n) {
  for (int i = blockIdx.x * blockDim.x + threadIdx.x; i < n;
       i += gridDim.x * blockDim.x)
    dst[i] = (__bf16)src[i];
}

__global__ void __launch_bounds__(256)
zero_f32(float* __restrict__ p, int n) {
  for (int i = blockIdx.x * blockDim.x + threadIdx.x; i < n;
       i += gridDim.x * blockDim.x)
    p[i] = 0.f;
}

// ---------------------------------------------------------------------------
// GEMM 1: qkv = x @ W_qkv^T. Block = 128x64 tile, 8 waves (4 along M x 2
// along N), wave = 32x32 (2x2 WMMA). Double-buffered async LDS staging.
// Epilogue: elu+1 on q,k; scatter into
//   qb  [bh, n, d], kbt [bh, d, n] (transposed), vbt [bh, e, n] (transposed)
// ---------------------------------------------------------------------------
__global__ void __launch_bounds__(256)
gemm_qkv(const __bf16* __restrict__ X, const __bf16* __restrict__ Wt,
         __bf16* __restrict__ qb, __bf16* __restrict__ kbt,
         __bf16* __restrict__ vbt) {
  __shared__ __align__(16) __bf16 As[2][BM * LDS_LD];
  __shared__ __align__(16) __bf16 Bs[2][BN * LDS_LD];

  const int tid = threadIdx.x, lane = tid & 31, wid = tid >> 5;
  const int wm = wid & 3, wn = wid >> 2;
  const int NTB = C3 / BN;                    // 36
  const int bn = blockIdx.x % NTB, bm = blockIdx.x / NTB;

  const __bf16* Ablk = X  + (size_t)bm * BM * C_DIM;
  const __bf16* Bblk = Wt + (size_t)bn * BN * C_DIM;

  stageA(As[0], Ablk, tid);
  stageB(Bs[0], Bblk, tid);

  v8f acc[2][2] = {};
  const int NSTEP = C_DIM / BK;               // 12
  for (int s = 0; s < NSTEP; ++s) {
    async_wait();
    __syncthreads();
    if (s + 1 < NSTEP) {
      stageA(As[(s + 1) & 1], Ablk + (s + 1) * BK, tid);
      stageB(Bs[(s + 1) & 1], Bblk + (s + 1) * BK, tid);
    }
    const __bf16* Ab = As[s & 1];
    const __bf16* Bb = Bs[s & 1];
    #pragma unroll
    for (int ks = 0; ks < BK; ks += 32) {
      v16bf a0 = frag_a(Ab + (wm * 32)      * LDS_LD + ks, LDS_LD, lane);
      v16bf a1 = frag_a(Ab + (wm * 32 + 16) * LDS_LD + ks, LDS_LD, lane);
      v16bf b0 = frag_b(Bb + (wn * 32)      * LDS_LD + ks, LDS_LD, lane);
      v16bf b1 = frag_b(Bb + (wn * 32 + 16) * LDS_LD + ks, LDS_LD, lane);
      WMMA_BF16(acc[0][0], a0, b0);
      WMMA_BF16(acc[0][1], a0, b1);
      WMMA_BF16(acc[1][0], a1, b0);
      WMMA_BF16(acc[1][1], a1, b1);
    }
  }

  const int hi8 = (lane >> 4) << 3;
  const int cn  = lane & 15;
  for (int ti = 0; ti < 2; ++ti)
    for (int tj = 0; tj < 2; ++tj)
      #pragma unroll
      for (int r = 0; r < 8; ++r) {
        const int m = bm * BM + wm * 32 + ti * 16 + hi8 + r;
        const int j = bn * BN + wn * 32 + tj * 16 + cn;
        const float val = acc[ti][tj][r];
        const int s   = j / C_DIM;             // 0:q 1:k 2:v
        const int rem = j - s * C_DIM;
        const int h = rem >> 6, dd = rem & 63;
        const int b = m >> 13, n = m & (N_SEQ - 1);
        const int bh = b * NUM_HEADS + h;
        if (s == 0)
          qb [(bh * N_SEQ + n) * 64 + dd] = (__bf16)elu1(val);
        else if (s == 1)
          kbt[(bh * 64 + dd) * N_SEQ + n] = (__bf16)elu1(val);
        else
          vbt[(bh * 64 + dd) * N_SEQ + n] = (__bf16)val;
      }
}

// ---------------------------------------------------------------------------
// GEMM 2: kvT[bh][e][d] = sum_n v[n,e]*k[n,d]. Wave = 32x32 (2x2 WMMA) so
// each k/v row is only read twice; K = 8192 split 16 ways, f32 atomic add.
// ---------------------------------------------------------------------------
__global__ void __launch_bounds__(256)
gemm_kv(const __bf16* __restrict__ vbt, const __bf16* __restrict__ kbt,
        float* __restrict__ kvT) {
  const int lane = threadIdx.x & 31;
  int task = blockIdx.x * 8 + (threadIdx.x >> 5);
  const int kc  = task & 15; task >>= 4;
  const int td2 = task & 1;  task >>= 1;
  const int te2 = task & 1;  task >>= 1;
  const int bh  = task;
  if (bh >= BH) return;

  const __bf16* A  = vbt + ((size_t)(bh * 64 + te2 * 32)) * N_SEQ + kc * 512;
  const __bf16* Bt = kbt + ((size_t)(bh * 64 + td2 * 32)) * N_SEQ + kc * 512;

  v8f acc[2][2] = {};
  for (int k0 = 0; k0 < 512; k0 += 32) {
    v16bf a0 = frag_a(A + k0,              N_SEQ, lane);
    v16bf a1 = frag_a(A + 16 * N_SEQ + k0, N_SEQ, lane);
    v16bf b0 = frag_b(Bt + k0,              N_SEQ, lane);
    v16bf b1 = frag_b(Bt + 16 * N_SEQ + k0, N_SEQ, lane);
    WMMA_BF16(acc[0][0], a0, b0);
    WMMA_BF16(acc[0][1], a0, b1);
    WMMA_BF16(acc[1][0], a1, b0);
    WMMA_BF16(acc[1][1], a1, b1);
  }

  const int hi8 = (lane >> 4) << 3;
  const int cn  = lane & 15;
  for (int ti = 0; ti < 2; ++ti)
    for (int tj = 0; tj < 2; ++tj)
      #pragma unroll
      for (int r = 0; r < 8; ++r) {
        const int e = te2 * 32 + ti * 16 + hi8 + r;
        const int d = td2 * 32 + tj * 16 + cn;
        atomicAdd(&kvT[bh * 4096 + e * 64 + d], acc[ti][tj][r]);
      }
}

// ---------------------------------------------------------------------------
// k_sum[bh][d] = sum_n k[n,d]   (one wave per row of kbt)
// ---------------------------------------------------------------------------
__global__ void __launch_bounds__(256)
ksum_kernel(const __bf16* __restrict__ kbt, float* __restrict__ ksum) {
  const int lane = threadIdx.x & 31;
  const int row  = blockIdx.x * 8 + (threadIdx.x >> 5);   // bh*64 + d
  if (row >= BH * 64) return;
  const __bf16* p = kbt + (size_t)row * N_SEQ;
  float s = 0.f;
  for (int i = lane * 8; i < N_SEQ; i += 32 * 8) {
    Pack8 u; u.q = *(const uint4*)(p + i);
    #pragma unroll
    for (int t = 0; t < 8; ++t) s += (float)u.h[t];
  }
  #pragma unroll
  for (int off = 16; off > 0; off >>= 1) s += __shfl_down(s, off, 32);
  if (lane == 0) ksum[row] = s;
}

// ---------------------------------------------------------------------------
// den[bh][n] = sum_d q[n,d] * ksum[d]
// ---------------------------------------------------------------------------
__global__ void __launch_bounds__(256)
den_kernel(const __bf16* __restrict__ qb, const float* __restrict__ ksum,
           float* __restrict__ den) {
  const int idx = blockIdx.x * blockDim.x + threadIdx.x;  // bh*N_SEQ + n
  if (idx >= BH * N_SEQ) return;
  const int bh = idx >> 13;
  const __bf16* qrow = qb + (size_t)idx * 64;
  const float*  ks   = ksum + bh * 64;
  float s = 0.f;
  #pragma unroll
  for (int d = 0; d < 64; ++d) s += (float)qrow[d] * ks[d];
  den[idx] = s;
}

// ---------------------------------------------------------------------------
// GEMM 3: numerator = q @ kv, normalized; wave does 16 rows x 64 cols,
// K = 64 in 2 WMMA steps. Bt = kvTb[bh][e][d] (K-contiguous, L2 resident).
// ---------------------------------------------------------------------------
__global__ void __launch_bounds__(256)
gemm_attn(const __bf16* __restrict__ qb, const __bf16* __restrict__ kvTb,
          const float* __restrict__ den, __bf16* __restrict__ attnb) {
  const int lane = threadIdx.x & 31;
  const int task = blockIdx.x * 8 + (threadIdx.x >> 5);
  const int nt = task & 511;
  const int bh = task >> 9;
  if (bh >= BH) return;

  const __bf16* A  = qb + ((size_t)bh * N_SEQ + nt * 16) * 64;
  const __bf16* Bt = kvTb + (size_t)bh * 4096;

  v8f acc[4] = {};
  #pragma unroll
  for (int k0 = 0; k0 < 64; k0 += 32) {
    v16bf a = frag_a(A + k0, 64, lane);
    #pragma unroll
    for (int te = 0; te < 4; ++te) {
      v16bf b = frag_b(Bt + te * 16 * 64 + k0, 64, lane);
      WMMA_BF16(acc[te], a, b);
    }
  }

  const int hi8 = (lane >> 4) << 3;
  const int cn  = lane & 15;
  const int b = bh / NUM_HEADS, h = bh % NUM_HEADS;
  #pragma unroll
  for (int r = 0; r < 8; ++r) {
    const int n = nt * 16 + hi8 + r;
    const float dinv = 1.f / (den[bh * N_SEQ + n] + 1e-6f);
    #pragma unroll
    for (int te = 0; te < 4; ++te) {
      const int e = te * 16 + cn;
      attnb[((size_t)(b * N_SEQ + n)) * C_DIM + h * 64 + e] =
          (__bf16)(acc[te][r] * dinv);
    }
  }
}

// ---------------------------------------------------------------------------
// GEMM 4: out = attn @ W_proj^T + b_proj, fp32 output. Same staged scheme.
// ---------------------------------------------------------------------------
__global__ void __launch_bounds__(256)
gemm_proj(const __bf16* __restrict__ Am, const __bf16* __restrict__ Wt,
          const float* __restrict__ bias, float* __restrict__ out) {
  __shared__ __align__(16) __bf16 As[2][BM * LDS_LD];
  __shared__ __align__(16) __bf16 Bs[2][BN * LDS_LD];

  const int tid = threadIdx.x, lane = tid & 31, wid = tid >> 5;
  const int wm = wid & 3, wn = wid >> 2;
  const int NTB = C_DIM / BN;                 // 12
  const int bn = blockIdx.x % NTB, bm = blockIdx.x / NTB;

  const __bf16* Ablk = Am + (size_t)bm * BM * C_DIM;
  const __bf16* Bblk = Wt + (size_t)bn * BN * C_DIM;

  stageA(As[0], Ablk, tid);
  stageB(Bs[0], Bblk, tid);

  v8f acc[2][2] = {};
  const int NSTEP = C_DIM / BK;               // 12
  for (int s = 0; s < NSTEP; ++s) {
    async_wait();
    __syncthreads();
    if (s + 1 < NSTEP) {
      stageA(As[(s + 1) & 1], Ablk + (s + 1) * BK, tid);
      stageB(Bs[(s + 1) & 1], Bblk + (s + 1) * BK, tid);
    }
    const __bf16* Ab = As[s & 1];
    const __bf16* Bb = Bs[s & 1];
    #pragma unroll
    for (int ks = 0; ks < BK; ks += 32) {
      v16bf a0 = frag_a(Ab + (wm * 32)      * LDS_LD + ks, LDS_LD, lane);
      v16bf a1 = frag_a(Ab + (wm * 32 + 16) * LDS_LD + ks, LDS_LD, lane);
      v16bf b0 = frag_b(Bb + (wn * 32)      * LDS_LD + ks, LDS_LD, lane);
      v16bf b1 = frag_b(Bb + (wn * 32 + 16) * LDS_LD + ks, LDS_LD, lane);
      WMMA_BF16(acc[0][0], a0, b0);
      WMMA_BF16(acc[0][1], a0, b1);
      WMMA_BF16(acc[1][0], a1, b0);
      WMMA_BF16(acc[1][1], a1, b1);
    }
  }

  const int hi8 = (lane >> 4) << 3;
  const int cn  = lane & 15;
  for (int ti = 0; ti < 2; ++ti)
    for (int tj = 0; tj < 2; ++tj) {
      const int jb = bn * BN + wn * 32 + tj * 16 + cn;
      const float bj = bias[jb];
      #pragma unroll
      for (int r = 0; r < 8; ++r) {
        const int m = bm * BM + wm * 32 + ti * 16 + hi8 + r;
        out[(size_t)m * C_DIM + jb] = acc[ti][tj][r] + bj;
      }
    }
}

// ---------------------------------------------------------------------------
extern "C" void kernel_launch(void* const* d_in, const int* in_sizes, int n_in,
                              void* d_out, int out_size, void* d_ws,
                              size_t ws_size, hipStream_t stream) {
  (void)in_sizes; (void)n_in; (void)out_size; (void)ws_size;
  const float* x     = (const float*)d_in[0];
  const float* Wqkv  = (const float*)d_in[1];
  const float* Wproj = (const float*)d_in[2];
  const float* bproj = (const float*)d_in[3];
  float* out = (float*)d_out;

  char* ws = (char*)d_ws;
  size_t off = 0;
  auto alloc = [&](size_t bytes) -> void* {
    void* p = ws + off;
    off = (off + bytes + 255) & ~(size_t)255;
    return p;
  };

  __bf16* xb     = (__bf16*)alloc((size_t)M_TOT * C_DIM * 2);
  __bf16* Wqkvb  = (__bf16*)alloc((size_t)C3 * C_DIM * 2);
  __bf16* Wprojb = (__bf16*)alloc((size_t)C_DIM * C_DIM * 2);
  __bf16* qb     = (__bf16*)alloc((size_t)BH * N_SEQ * 64 * 2);
  __bf16* kbt    = (__bf16*)alloc((size_t)BH * 64 * N_SEQ * 2);
  __bf16* vbt    = (__bf16*)alloc((size_t)BH * 64 * N_SEQ * 2);
  float*  kvT    = (float*)alloc((size_t)BH * 4096 * 4);
  __bf16* kvTb   = (__bf16*)alloc((size_t)BH * 4096 * 2);
  float*  ksum   = (float*)alloc((size_t)BH * 64 * 4);
  float*  den    = (float*)alloc((size_t)BH * N_SEQ * 4);
  __bf16* attnb  = (__bf16*)alloc((size_t)M_TOT * C_DIM * 2);

  cvt_f32_bf16<<<4096, 256, 0, stream>>>(x, xb, M_TOT * C_DIM);
  cvt_f32_bf16<<<512,  256, 0, stream>>>(Wqkv, Wqkvb, C3 * C_DIM);
  cvt_f32_bf16<<<256,  256, 0, stream>>>(Wproj, Wprojb, C_DIM * C_DIM);

  gemm_qkv<<<(M_TOT / BM) * (C3 / BN), 256, 0, stream>>>(xb, Wqkvb, qb, kbt,
                                                         vbt);

  zero_f32<<<192, 256, 0, stream>>>(kvT, BH * 4096);
  gemm_kv<<<BH * 4 * 16 / 8, 256, 0, stream>>>(vbt, kbt, kvT);

  ksum_kernel<<<BH * 64 / 8, 256, 0, stream>>>(kbt, ksum);
  den_kernel<<<BH * N_SEQ / 256, 256, 0, stream>>>(qb, ksum, den);
  cvt_f32_bf16<<<192, 256, 0, stream>>>(kvT, kvTb, BH * 4096);

  gemm_attn<<<BH * 512 / 8, 256, 0, stream>>>(qb, kvTb, den, attnb);

  gemm_proj<<<(M_TOT / BM) * (C_DIM / BN), 256, 0, stream>>>(attnb, Wprojb,
                                                             bproj, out);
}
